// FCNN_76261439308466
// MI455X (gfx1250) — matrix operand order repaired and checked by reading, not compiled, not run on hardware
//
#include <hip/hip_runtime.h>
#include <hip/hip_bf16.h>

// ---------------------------------------------------------------------------
// Patch-MLP (243 -> 50 -> 50 -> 9) over 541,696 positions, fused via
// v_wmma_f32_16x16x32_f16, followed by gather-style fold.
// ---------------------------------------------------------------------------

typedef _Float16 v16h __attribute__((ext_vector_type(16)));
typedef _Float16 v8h  __attribute__((ext_vector_type(8)));
typedef float    v8f  __attribute__((ext_vector_type(8)));

union V16U { v16h v; v8h h[2]; };

static __device__ inline v16h frag2(const _Float16* p0, const _Float16* p1) {
    V16U u;
    u.h[0] = *(const v8h*)p0;   // 16B aligned -> b128
    u.h[1] = *(const v8h*)p1;
    return u.v;
}

constexpr int kB2   = 16;            // B*L
constexpr int kHp   = 184;           // unfold positions per axis
constexpr int kWp   = 184;
constexpr int kHW   = kHp * kWp;     // 33856
constexpr int kNPos = kB2 * kHW;     // 541696
constexpr int kIH   = 192;
constexpr int kIW   = 192;
constexpr int kCin  = 3;
constexpr int kK1   = 243;           // C*9*9
constexpr int kK1P  = 256;           // padded
constexpr int kWid  = 50;
constexpr int kNP   = 64;            // padded hidden width
constexpr int kOutD = 9;
constexpr int kOH   = 188;
constexpr int kOW   = 188;

constexpr int kWaves = 4;            // waves per block
constexpr int kTile  = 16;           // positions per wave

// ---------------------------------------------------------------------------
// Weight prep: fp32 -> f16, layout [n][k] (row-major, k contiguous), zero pad.
// W1t: 64x256, W2t: 64x64, W3t: 16x64.
// ---------------------------------------------------------------------------
__global__ void prep_weights(const float* __restrict__ W1,
                             const float* __restrict__ W2,
                             const float* __restrict__ W3,
                             _Float16* __restrict__ W1t,
                             _Float16* __restrict__ W2t,
                             _Float16* __restrict__ W3t) {
    int i = blockIdx.x * blockDim.x + threadIdx.x;
    if (i < 64 * 256) {
        int n = i >> 8, k = i & 255;
        W1t[i] = (n < kWid && k < kK1) ? (_Float16)W1[n * kK1 + k] : (_Float16)0.f;
    } else if (i < 64 * 256 + 64 * 64) {
        int j = i - 64 * 256;
        int n = j >> 6, k = j & 63;
        W2t[j] = (n < kWid && k < kWid) ? (_Float16)W2[n * kWid + k] : (_Float16)0.f;
    } else if (i < 64 * 256 + 64 * 64 + 16 * 64) {
        int j = i - (64 * 256 + 64 * 64);
        int n = j >> 6, k = j & 63;
        W3t[j] = (n < kOutD && k < kWid) ? (_Float16)W3[n * kWid + k] : (_Float16)0.f;
    }
}

// ---------------------------------------------------------------------------
// Fused 3-layer MLP; one wave handles a 16-position tile.
// ---------------------------------------------------------------------------
__global__ __launch_bounds__(kWaves * 32)
void mlp_kernel(const float* __restrict__ x,
                const float* __restrict__ b1,
                const float* __restrict__ b2v,
                const float* __restrict__ b3,
                const _Float16* __restrict__ W1t,
                const _Float16* __restrict__ W2t,
                const _Float16* __restrict__ W3t,
                float* __restrict__ y) {   // [kNPos][9]
    // row strides are multiples of 16B and coprime-ish with 64 banks
    __shared__ __align__(16) _Float16 sFeat[kWaves][kTile][264]; // 33,792 B
    __shared__ __align__(16) _Float16 sH1[kWaves][kTile][72];    //  9,216 B
    __shared__ __align__(16) _Float16 sH2[kWaves][kTile][72];    //  9,216 B

    const int lane = threadIdx.x & 31;
    const int wave = threadIdx.x >> 5;
    const int tile = blockIdx.x * kWaves + wave;
    const int n0   = tile * kTile;

    // ---- stage features: 16 positions x 243 feats, fp32 -> f16, into LDS ----
    for (int r = lane; r < kTile * 27; r += 32) {
        int p  = r / 27;
        int cr = r - p * 27;
        int c  = cr / 9;
        int kh = cr - c * 9;
        int n  = n0 + p;
        int b  = n / kHW;
        int rm = n - b * kHW;
        int ph = rm / kWp;
        int pw = rm - ph * kWp;
        const float* src = x + ((b * kCin + c) * kIH + (ph + kh)) * kIW + pw;
        _Float16* dst = &sFeat[wave][p][c * 81 + kh * 9];
        #pragma unroll
        for (int j = 0; j < 9; ++j) dst[j] = (_Float16)src[j];
    }
    // zero the k = 243..255 padding (k >= 256 is never read)
    for (int r = lane; r < kTile * 13; r += 32) {
        int p = r / 13;
        sFeat[wave][p][kK1 + (r - p * 13)] = (_Float16)0.f;
    }
    __syncthreads();

    const int hs   = (lane < 16) ? 0 : 1;   // lane half-group
    const int mcol = lane & 15;

    // ---- layer 1: [16x256] x [256x64], ReLU, -> sH1 ----
    #pragma unroll
    for (int nt = 0; nt < 4; ++nt) {
        const int nb = nt * 16 + mcol;
        const float bias = (nb < kWid) ? b1[nb] : 0.f;
        v8f acc;
        #pragma unroll
        for (int r = 0; r < 8; ++r) acc[r] = bias;
        #pragma unroll
        for (int kc = 0; kc < 8; ++kc) {
            const int ka = kc * 32 + hs * 8;
            v16h a = frag2(&sFeat[wave][mcol][ka], &sFeat[wave][mcol][ka + 16]);
            const _Float16* wr = W1t + nb * kK1P + kc * 32 + hs * 8;
            v16h bf = frag2(wr, wr + 16);
            acc = __builtin_amdgcn_wmma_f32_16x16x32_f16(
                      false, a, false, bf, (short)0, acc, false, false);
        }
        #pragma unroll
        for (int r = 0; r < 8; ++r) {
            float v = acc[r] > 0.f ? acc[r] : 0.f;
            sH1[wave][r + 8 * hs][nt * 16 + mcol] = (_Float16)v;
        }
    }
    __syncthreads();

    // ---- layer 2: [16x64] x [64x64], ReLU, -> sH2 ----
    #pragma unroll
    for (int nt = 0; nt < 4; ++nt) {
        const int nb = nt * 16 + mcol;
        const float bias = (nb < kWid) ? b2v[nb] : 0.f;
        v8f acc;
        #pragma unroll
        for (int r = 0; r < 8; ++r) acc[r] = bias;
        #pragma unroll
        for (int kc = 0; kc < 2; ++kc) {
            const int ka = kc * 32 + hs * 8;
            v16h a = frag2(&sH1[wave][mcol][ka], &sH1[wave][mcol][ka + 16]);
            const _Float16* wr = W2t + nb * kNP + kc * 32 + hs * 8;
            v16h bf = frag2(wr, wr + 16);
            acc = __builtin_amdgcn_wmma_f32_16x16x32_f16(
                      false, a, false, bf, (short)0, acc, false, false);
        }
        #pragma unroll
        for (int r = 0; r < 8; ++r) {
            float v = acc[r] > 0.f ? acc[r] : 0.f;
            sH2[wave][r + 8 * hs][nt * 16 + mcol] = (_Float16)v;
        }
    }
    __syncthreads();

    // ---- layer 3: [16x64] x [64x16], write y[n][0..8] ----
    {
        const int nb = mcol;
        const float bias = (nb < kOutD) ? b3[nb] : 0.f;
        v8f acc;
        #pragma unroll
        for (int r = 0; r < 8; ++r) acc[r] = bias;
        #pragma unroll
        for (int kc = 0; kc < 2; ++kc) {
            const int ka = kc * 32 + hs * 8;
            v16h a = frag2(&sH2[wave][mcol][ka], &sH2[wave][mcol][ka + 16]);
            const _Float16* wr = W3t + nb * kNP + kc * 32 + hs * 8;
            v16h bf = frag2(wr, wr + 16);
            acc = __builtin_amdgcn_wmma_f32_16x16x32_f16(
                      false, a, false, bf, (short)0, acc, false, false);
        }
        if (mcol < kOutD) {
            #pragma unroll
            for (int r = 0; r < 8; ++r) {
                const int m = r + 8 * hs;
                y[(long)(n0 + m) * kOutD + mcol] = acc[r];
            }
        }
    }
}

// ---------------------------------------------------------------------------
// Fold: each output pixel gathers its <=9 patch contributions (no atomics).
// ---------------------------------------------------------------------------
__global__ void fold_kernel(const float* __restrict__ y, float* __restrict__ out) {
    int idx = blockIdx.x * blockDim.x + threadIdx.x;
    if (idx >= kB2 * kOH * kOW) return;
    int b  = idx / (kOH * kOW);
    int rm = idx - b * (kOH * kOW);
    int oh = rm / kOW;
    int ow = rm - oh * kOW;
    float sum = 0.f;
    int cnt = 0;
    #pragma unroll
    for (int di = 0; di < 3; ++di) {
        int ph = oh - di;
        if (ph < 0 || ph >= kHp) continue;
        #pragma unroll
        for (int dj = 0; dj < 3; ++dj) {
            int pw = ow - dj;
            if (pw < 0 || pw >= kWp) continue;
            sum += y[((long)b * kHW + ph * kWp + pw) * kOutD + di * 3 + dj];
            ++cnt;
        }
    }
    out[idx] = sum / (float)cnt;
}

// ---------------------------------------------------------------------------
extern "C" void kernel_launch(void* const* d_in, const int* in_sizes, int n_in,
                              void* d_out, int out_size, void* d_ws, size_t ws_size,
                              hipStream_t stream) {
    const float* x  = (const float*)d_in[0];
    const float* W1 = (const float*)d_in[1];
    const float* b1 = (const float*)d_in[2];
    const float* W2 = (const float*)d_in[3];
    const float* b2 = (const float*)d_in[4];
    const float* W3 = (const float*)d_in[5];
    const float* b3 = (const float*)d_in[6];
    float* out = (float*)d_out;

    char* ws = (char*)d_ws;
    float* yBuf = (float*)ws;                           // kNPos*9 f32 = 19,501,056 B
    size_t off = (size_t)kNPos * kOutD * sizeof(float); // 16B aligned
    _Float16* W1t = (_Float16*)(ws + off); off += 64 * 256 * sizeof(_Float16);
    _Float16* W2t = (_Float16*)(ws + off); off += 64 * 64  * sizeof(_Float16);
    _Float16* W3t = (_Float16*)(ws + off); off += 16 * 64  * sizeof(_Float16);

    const int prepN = 64 * 256 + 64 * 64 + 16 * 64;     // 21504
    prep_weights<<<(prepN + 255) / 256, 256, 0, stream>>>(W1, W2, W3, W1t, W2t, W3t);

    const int tiles  = kNPos / kTile;                   // 33856 (exact)
    const int blocks = tiles / kWaves;                  // 8464  (exact)
    mlp_kernel<<<blocks, kWaves * 32, 0, stream>>>(x, b1, b2, b3, W1t, W2t, W3t, yBuf);

    const int nOut = kB2 * kOH * kOW;                   // 565504
    fold_kernel<<<(nOut + 255) / 256, 256, 0, stream>>>(yBuf, out);
}
